// VanillaRNN_50105088475253
// MI455X (gfx1250) — matrix operand order repaired and checked by reading, not compiled, not run on hardware
//
#include <hip/hip_runtime.h>

// ---------------------------------------------------------------------------
// Vanilla RNN step on MI455X (gfx1250, wave32, WMMA f16, f32 accumulate).
//   NS  = ReLU(X @ W_in^T + state @ W_rec^T + b_in + b_rec)   [8192 x 4096]
//   out = ReLU(NS @ W_out^T + b_out)                          [8192 x 1024]
// d_out = [ out | NS ] (f32, reference return order).
//
// v2: software-pipelined, LDS double-buffered GEMM.
//   - BK=64: 16 v_wmma per iteration, one barrier per iteration
//   - global prefetch of tile i+1 held in raw float4 regs across the WMMA
//     phase of tile i; f32->f16 conversion deferred to the LDS store phase
//   - the two fused GEMMs (X,W_in) and (state,W_rec) form one iteration space
// ---------------------------------------------------------------------------

typedef _Float16 v8h  __attribute__((ext_vector_type(8)));
typedef _Float16 v16h __attribute__((ext_vector_type(16)));
typedef float    v8f  __attribute__((ext_vector_type(8)));

#define BM   128
#define BN   128
#define BK   64
#define LDSS 72   // halves per LDS row: 64 data + 8 pad (144B rows, 16B aligned)

// Load one 16-float row-chunk (task = row*4 + quarter) into 4 float4 regs.
static __device__ __forceinline__ void load_chunk(
    float4* __restrict__ dst, const float* __restrict__ src,
    int ld, int row0, int task, int k0)
{
    const int row = task >> 2;          // 0..127
    const int q   = task & 3;           // 0..3 -> cols q*16 .. q*16+15
    const float* p = src + (size_t)(row0 + row) * (size_t)ld + (k0 + q * 16);
    dst[0] = ((const float4*)p)[0];
    dst[1] = ((const float4*)p)[1];
    dst[2] = ((const float4*)p)[2];
    dst[3] = ((const float4*)p)[3];
}

// Convert 16 f32 -> 16 f16 and store to the padded LDS tile.
static __device__ __forceinline__ void cvt_store_chunk(
    _Float16* __restrict__ lds, int task, const float4* __restrict__ f)
{
    const int row = task >> 2;
    const int q   = task & 3;
    v8h h0 = { (_Float16)f[0].x, (_Float16)f[0].y, (_Float16)f[0].z, (_Float16)f[0].w,
               (_Float16)f[1].x, (_Float16)f[1].y, (_Float16)f[1].z, (_Float16)f[1].w };
    v8h h1 = { (_Float16)f[2].x, (_Float16)f[2].y, (_Float16)f[2].z, (_Float16)f[2].w,
               (_Float16)f[3].x, (_Float16)f[3].y, (_Float16)f[3].z, (_Float16)f[3].w };
    _Float16* p = lds + row * LDSS + q * 16;
    *(v8h*)(p + 0) = h0;
    *(v8h*)(p + 8) = h1;
}

// A-matrix 16x32 f16 fragment (CDNA5 layout), tile column offset kk:
//   lane<16 : M=lane,    halves 0..7 -> K=0..7,  halves 8..15 -> K=16..23
//   lane>=16: M=lane-16, halves 0..7 -> K=8..15, halves 8..15 -> K=24..31
static __device__ __forceinline__ v16h load_a_frag(
    const _Float16* __restrict__ As, int mbase, int kk, int lane)
{
    const int m   = mbase + (lane & 15);
    const int klo = (lane >> 4) << 3;          // 0 or 8
    const _Float16* r = As + m * LDSS + kk;
    union { v16h v; v8h h[2]; } u;
    u.h[0] = *(const v8h*)(r + klo);
    u.h[1] = *(const v8h*)(r + klo + 16);
    return u.v;
}

// B-matrix 32x16 f16 fragment: lane holds column N=lane%16,
// K = kk+0..7,16..23 (lane<16) or kk+8..15,24..31 (lane>=16), same as A.
static __device__ __forceinline__ v16h load_b_frag(
    const _Float16* __restrict__ Bs, int nbase, int kk, int lane)
{
    const int n  = nbase + (lane & 15);
    const int kb = (lane >> 4) << 3;           // 0 or 8
    const _Float16* r = Bs + n * LDSS + kk + kb;
    union { v16h v; v8h h[2]; } u;
    u.h[0] = *(const v8h*)(r + 0);
    u.h[1] = *(const v8h*)(r + 16);
    return u.v;
}

__global__ __launch_bounds__(256)
void rnn_gemm_relu(const float* __restrict__ A1, const float* __restrict__ W1, int K1,
                   const float* __restrict__ A2, const float* __restrict__ W2, int K2,
                   const float* __restrict__ b1, const float* __restrict__ b2,
                   float* __restrict__ C, int N)
{
    __shared__ _Float16 As[2][BM * LDSS];   // 2 x 18 KB
    __shared__ _Float16 Bs[2][BN * LDSS];   // 2 x 18 KB

    const int tid    = threadIdx.x;
    const int lane   = tid & 31;
    const int wave   = tid >> 5;               // 8 waves
    const int waveM  = wave & 1;               // 2 waves in M (64 rows each)
    const int waveN  = wave >> 1;              // 4 waves in N (32 cols each)
    const int blockM = blockIdx.x * BM;
    const int blockN = blockIdx.y * BN;

    const v8f vzero = {0.f, 0.f, 0.f, 0.f, 0.f, 0.f, 0.f, 0.f};
    v8f acc[4][2];
#pragma unroll
    for (int i = 0; i < 4; ++i)
#pragma unroll
        for (int j = 0; j < 2; ++j)
            acc[i][j] = vzero;

    // Unified iteration space over both fused GEMM sources.
    const int it1 = K1 / BK;
    const int itn = it1 + ((A2 != nullptr) ? (K2 / BK) : 0);

    auto src_of = [&](int it, const float*& A, const float*& W, int& ld, int& k0) {
        if (it < it1) { A = A1; W = W1; ld = K1; k0 = it * BK; }
        else          { A = A2; W = W2; ld = K2; k0 = (it - it1) * BK; }
    };

    // Prefetch registers: 2 chunks of A + 2 chunks of B tile (16 floats each).
    float4 pf[16];

    // ---- prologue: tile 0 -> LDS buffer 0 ----
    {
        const float *A, *W; int ld, k0;
        src_of(0, A, W, ld, k0);
        load_chunk(pf + 0,  A, ld, blockM, tid,       k0);
        load_chunk(pf + 4,  A, ld, blockM, tid + 256, k0);
        load_chunk(pf + 8,  W, ld, blockN, tid,       k0);
        load_chunk(pf + 12, W, ld, blockN, tid + 256, k0);
        cvt_store_chunk(As[0], tid,       pf + 0);
        cvt_store_chunk(As[0], tid + 256, pf + 4);
        cvt_store_chunk(Bs[0], tid,       pf + 8);
        cvt_store_chunk(Bs[0], tid + 256, pf + 12);
    }
    __syncthreads();

    for (int it = 0; it < itn; ++it) {
        const int  cur     = it & 1;
        const bool hasNext = (it + 1 < itn);

        // Issue next tile's global loads early; conversion deferred so the
        // loadcnt wait lands after the WMMA phase below.
        if (hasNext) {
            const float *A, *W; int ld, k0;
            src_of(it + 1, A, W, ld, k0);
            load_chunk(pf + 0,  A, ld, blockM, tid,       k0);
            load_chunk(pf + 4,  A, ld, blockM, tid + 256, k0);
            load_chunk(pf + 8,  W, ld, blockN, tid,       k0);
            load_chunk(pf + 12, W, ld, blockN, tid + 256, k0);
        }

        // ---- WMMA phase on buffer `cur`: 2 k-halves x 8 = 16 WMMAs ----
        const _Float16* Asc = As[cur];
        const _Float16* Bsc = Bs[cur];
#pragma unroll
        for (int kk = 0; kk < BK; kk += 32) {
            v16h a[4], b[2];
#pragma unroll
            for (int i = 0; i < 4; ++i)
                a[i] = load_a_frag(Asc, waveM * 64 + i * 16, kk, lane);
#pragma unroll
            for (int j = 0; j < 2; ++j)
                b[j] = load_b_frag(Bsc, waveN * 32 + j * 16, kk, lane);
#pragma unroll
            for (int i = 0; i < 4; ++i)
#pragma unroll
                for (int j = 0; j < 2; ++j)
                    acc[i][j] = __builtin_amdgcn_wmma_f32_16x16x32_f16(
                        /*neg_a=*/false, a[i], /*neg_b=*/false, b[j],
                        /*c_mod=*/(short)0, acc[i][j],
                        /*reuse_a=*/false, /*reuse_b=*/false);
        }

        // ---- store prefetched tile into the other buffer ----
        if (hasNext) {
            const int nxt = cur ^ 1;
            cvt_store_chunk(As[nxt], tid,       pf + 0);
            cvt_store_chunk(As[nxt], tid + 256, pf + 4);
            cvt_store_chunk(Bs[nxt], tid,       pf + 8);
            cvt_store_chunk(Bs[nxt], tid + 256, pf + 12);
        }
        __syncthreads();   // one barrier per iteration (double-buffer safe)
    }

    // Epilogue: C/D layout -> lane holds N=lane%16; VGPR v -> M = v + 8*(lane>=16)
    const int col16   = lane & 15;
    const int row_off = (lane >> 4) * 8;
#pragma unroll
    for (int i = 0; i < 4; ++i) {
#pragma unroll
        for (int j = 0; j < 2; ++j) {
            const int gn = blockN + waveN * 32 + j * 16 + col16;
            float bias = b1[gn];
            if (b2 != nullptr) bias += b2[gn];
#pragma unroll
            for (int v = 0; v < 8; ++v) {
                const int gm = blockM + waveM * 64 + i * 16 + row_off + v;
                const float x = acc[i][j][v] + bias;
                C[(size_t)gm * (size_t)N + gn] = x > 0.f ? x : 0.f;
            }
        }
    }
}

extern "C" void kernel_launch(void* const* d_in, const int* in_sizes, int n_in,
                              void* d_out, int out_size, void* d_ws, size_t ws_size,
                              hipStream_t stream)
{
    (void)in_sizes; (void)n_in; (void)out_size; (void)d_ws; (void)ws_size;

    constexpr int Bsz  = 8192;
    constexpr int DIN  = 1024;
    constexpr int H    = 4096;
    constexpr int DOUT = 1024;

    const float* X     = (const float*)d_in[0];
    const float* state = (const float*)d_in[1];
    const float* W_in  = (const float*)d_in[2];
    const float* b_in  = (const float*)d_in[3];
    const float* W_rec = (const float*)d_in[4];
    const float* b_rec = (const float*)d_in[5];
    const float* W_out = (const float*)d_in[6];
    const float* b_out = (const float*)d_in[7];

    float* out = (float*)d_out;                       // [B, D_OUT]
    float* NS  = out + (size_t)Bsz * (size_t)DOUT;    // [B, H] (new_state)

    // Kernel 1: NS = ReLU(X @ W_in^T + state @ W_rec^T + b_in + b_rec)
    rnn_gemm_relu<<<dim3(Bsz / BM, H / BN), dim3(256), 0, stream>>>(
        X, W_in, DIN, state, W_rec, H, b_in, b_rec, NS, H);

    // Kernel 2: out = ReLU(NS @ W_out^T + b_out)  (stream-ordered after k1)
    rnn_gemm_relu<<<dim3(Bsz / BM, DOUT / BN), dim3(256), 0, stream>>>(
        NS, W_out, H, nullptr, nullptr, 0, b_out, nullptr, out, DOUT);
}
